// MultiheadAttention_80470507257862
// MI455X (gfx1250) — matrix operand order repaired and checked by reading, not compiled, not run on hardware
//
#include <hip/hip_runtime.h>
#include <hip/hip_bf16.h>
#include <stdint.h>

typedef __attribute__((ext_vector_type(16))) __bf16 v16bf;
typedef __attribute__((ext_vector_type(8)))  __bf16 v8bf;
typedef __attribute__((ext_vector_type(8)))  float  v8f;

#define BATCH 8
#define NTOK  448      // tokens (s and t)
#define DM    64       // d_model
#define NPOS  128      // 16*8 spatial positions in v
#define FDIM  8192     // NPOS*DM
#define PPB   4        // pos per block (one pos per wave)

__device__ __forceinline__ uint16_t f2bf(float f) {
  uint32_t u = __float_as_uint(f);
  u += 0x7FFFu + ((u >> 16) & 1u);          // round-to-nearest-even
  return (uint16_t)(u >> 16);
}

// ---------------------------------------------------------------------------
// Kernel 1: grouped conv projection. Block = one (b, token), 64 threads = d_model.
// Produces both branch projections (shared input reads).
__global__ void proj_kernel(const float* __restrict__ x,
                            const float* __restrict__ w0,
                            const float* __restrict__ w1,
                            float* __restrict__ out0,
                            float* __restrict__ out1) {
  __shared__ float s0[144], s1[144];
  int bt = blockIdx.x;               // b*448 + t
  int b = bt / NTOK, t = bt % NTOK;
  int c = t >> 6, ij = t & 63, i = ij >> 3, j = ij & 7;
  for (int idx = threadIdx.x; idx < 144; idx += 64) {
    s0[idx] = w0[c * 144 + idx];
    s1[idx] = w1[c * 144 + idx];
  }
  __syncthreads();
  int l = threadIdx.x;
  const float* base = x + ((((size_t)b * 7 + c) * 96 + i * 12) * 96 + j * 12) * 64 + l;
  float a0 = 0.f, a1 = 0.f;
  for (int p = 0; p < 12; ++p)
    for (int q = 0; q < 12; ++q) {
      float xv = base[(p * 96 + q) * 64];
      a0 += xv * s0[p * 12 + q];
      a1 += xv * s1[p * 12 + q];
    }
  out0[(size_t)bt * 64 + l] = a0;
  out1[(size_t)bt * 64 + l] = a1;
}

// ---------------------------------------------------------------------------
// Kernel 2: fold wv into fc:  M0[o,d'] = sum_d fc_w[o,d]*wv0[d,d'], M1 likewise.
__global__ void mproj_kernel(const float* __restrict__ fc_w,
                             const float* __restrict__ wv0,
                             const float* __restrict__ wv1,
                             float* __restrict__ M0, float* __restrict__ M1) {
  int gid = blockIdx.x * blockDim.x + threadIdx.x;   // 4096 threads
  int o = gid >> 6, dp = gid & 63;
  float m0 = 0.f, m1 = 0.f;
  for (int d = 0; d < 64; ++d) {
    m0 += fc_w[o * 128 + d]      * wv0[d * 64 + dp];
    m1 += fc_w[o * 128 + 64 + d] * wv1[d * 64 + dp];
  }
  M0[o * 64 + dp] = m0;
  M1[o * 64 + dp] = m1;
}

// ---------------------------------------------------------------------------
// Kernel 3: transpose+pack v (f32, [b][t][f]) -> vt (bf16, [b][f][t]).
// 32x32 LDS tile transpose; block (32,8), each thread handles 4 elements.
__global__ void vpack_kernel(const float* __restrict__ v, uint16_t* __restrict__ vt) {
  __shared__ float tile[32][33];
  int f0 = blockIdx.x * 32, t0 = blockIdx.y * 32, b = blockIdx.z;
  int tx = threadIdx.x, ty = threadIdx.y;
  for (int r = 0; r < 4; ++r) {
    int t = t0 + ty + 8 * r;
    tile[ty + 8 * r][tx] = v[((size_t)b * NTOK + t) * FDIM + f0 + tx];
  }
  __syncthreads();
  for (int r = 0; r < 4; ++r) {
    int f = f0 + ty + 8 * r;
    vt[((size_t)b * FDIM + f) * NTOK + t0 + tx] = f2bf(tile[tx][ty + 8 * r]);
  }
}

// ---------------------------------------------------------------------------
// Kernel 4: attention logits + mask + softmax for both branches; output bf16.
// Block = one (b,s) row, 256 threads over t.
__global__ void scores_kernel(const float* __restrict__ q00, const float* __restrict__ q01,
                              const float* __restrict__ k00, const float* __restrict__ k01,
                              const int* __restrict__ mask,
                              uint16_t* __restrict__ a0, uint16_t* __restrict__ a1) {
  __shared__ float qr0[64], qr1[64];
  __shared__ float l0[NTOK], l1[NTOK];
  __shared__ float red0[256], red1[256];
  int bs = blockIdx.x;
  int b = bs / NTOK;
  int tid = threadIdx.x;
  if (tid < 64) {
    qr0[tid] = q00[(size_t)bs * 64 + tid];
    qr1[tid] = q01[(size_t)bs * 64 + tid];
  }
  __syncthreads();
  const int* mrow = mask + (size_t)bs * NTOK;
  float lmax0 = -3e38f, lmax1 = -3e38f;
  for (int t = tid; t < NTOK; t += 256) {
    const float* kr0 = k00 + ((size_t)b * NTOK + t) * 64;
    const float* kr1 = k01 + ((size_t)b * NTOK + t) * 64;
    float d0 = 0.f, d1 = 0.f;
    for (int l = 0; l < 64; ++l) { d0 += qr0[l] * kr0[l]; d1 += qr1[l] * kr1[l]; }
    d0 *= 0.125f; d1 *= 0.125f;                 // 1/sqrt(64)
    if (mrow[t] == 0) { d0 = -1e9f; d1 = -1e9f; }
    l0[t] = d0; l1[t] = d1;
    lmax0 = fmaxf(lmax0, d0); lmax1 = fmaxf(lmax1, d1);
  }
  red0[tid] = lmax0; red1[tid] = lmax1;
  __syncthreads();
  for (int off = 128; off > 0; off >>= 1) {
    if (tid < off) {
      red0[tid] = fmaxf(red0[tid], red0[tid + off]);
      red1[tid] = fmaxf(red1[tid], red1[tid + off]);
    }
    __syncthreads();
  }
  float m0 = red0[0], m1 = red1[0];
  __syncthreads();
  float ls0 = 0.f, ls1 = 0.f;
  for (int t = tid; t < NTOK; t += 256) {
    float e0 = __expf(l0[t] - m0), e1 = __expf(l1[t] - m1);
    l0[t] = e0; l1[t] = e1;
    ls0 += e0; ls1 += e1;
  }
  red0[tid] = ls0; red1[tid] = ls1;
  __syncthreads();
  for (int off = 128; off > 0; off >>= 1) {
    if (tid < off) { red0[tid] += red0[tid + off]; red1[tid] += red1[tid + off]; }
    __syncthreads();
  }
  float inv0 = 1.f / red0[0], inv1 = 1.f / red1[0];
  for (int t = tid; t < NTOK; t += 256) {
    a0[(size_t)bs * NTOK + t] = f2bf(l0[t] * inv0);
    a1[(size_t)bs * NTOK + t] = f2bf(l1[t] * inv1);
  }
}

// ---------------------------------------------------------------------------
// Kernel 5: main GEMM  av = a @ v  via bf16 WMMA, fused epilogue (M0/M1 + bias).
// Block = 4 waves = one (b, 16-row s-tile, 4 pos); wave wv owns pos0+wv entirely
// (4 B tiles). One A-fragment pair feeds 8 WMMAs -> 1.5 b128 loads per WMMA.
__global__ void __launch_bounds__(128) attnmm_kernel(
    const uint16_t* __restrict__ a0, const uint16_t* __restrict__ a1,
    const uint16_t* __restrict__ vt,
    const float* __restrict__ M0, const float* __restrict__ M1,
    const float* __restrict__ fcb, float* __restrict__ out) {
  __shared__ float av0s[16][PPB * 64 + 1];   // odd stride -> conflict-free
  __shared__ float av1s[16][PPB * 64 + 1];
  int st = blockIdx.x, pos0 = blockIdx.y * PPB, b = blockIdx.z;
  int tid  = threadIdx.x;
  int wv   = tid >> 5;           // wave id 0..3 -> pos = pos0 + wv
  int lane = tid & 31;
  int kh   = lane >> 4;          // lane half selects K sub-range
  int cl   = lane & 15;          // row (A) / column (B) within tile
  // A fragment (16-bit 16x32): lanes<16 hold K 0-7 & 16-23; lanes>=16 hold 8-15 & 24-31
  const uint16_t* arow0 = a0 + ((size_t)b * NTOK + st * 16 + cl) * NTOK + kh * 8;
  const uint16_t* arow1 = a1 + ((size_t)b * NTOK + st * 16 + cl) * NTOK + kh * 8;
  // B fragments (16-bit 32x16): lane = column; lanes<16 hold K 0-7 & 16-23; lanes>=16 hold 8-15 & 24-31.
  // Tiles j=0..3 are at immediate offsets j*16*NTOK*2 bytes from one base.
  const uint16_t* bcol =
      vt + ((size_t)b * FDIM + (pos0 + wv) * 64 + cl) * NTOK + kh * 8;
  v8f acc0[PPB], acc1[PPB];
#pragma unroll
  for (int j = 0; j < PPB; ++j) {
    acc0[j] = (v8f){0.f, 0.f, 0.f, 0.f, 0.f, 0.f, 0.f, 0.f};
    acc1[j] = (v8f){0.f, 0.f, 0.f, 0.f, 0.f, 0.f, 0.f, 0.f};
  }
  for (int k0 = 0; k0 < NTOK; k0 += 32) {      // 14 K-steps exactly
    union { v16bf v; v8bf h[2]; } fa0, fa1;
    fa0.h[0] = *reinterpret_cast<const v8bf*>(arow0 + k0);
    fa0.h[1] = *reinterpret_cast<const v8bf*>(arow0 + k0 + 16);
    fa1.h[0] = *reinterpret_cast<const v8bf*>(arow1 + k0);
    fa1.h[1] = *reinterpret_cast<const v8bf*>(arow1 + k0 + 16);
    __builtin_prefetch(bcol + k0 + 224, 0, 0);  // global_prefetch_b8 ahead on B
#pragma unroll
    for (int j = 0; j < PPB; ++j) {
      union { v16bf v; v8bf h[2]; } fbu;
      fbu.h[0] = *reinterpret_cast<const v8bf*>(bcol + (size_t)j * 16 * NTOK + k0);
      fbu.h[1] = *reinterpret_cast<const v8bf*>(bcol + (size_t)j * 16 * NTOK + k0 + 16);
      v16bf fb = fbu.v;
      acc0[j] = __builtin_amdgcn_wmma_f32_16x16x32_bf16(
          false, fa0.v, false, fb, (short)0, acc0[j], false, false);
      acc1[j] = __builtin_amdgcn_wmma_f32_16x16x32_bf16(
          false, fa1.v, false, fb, (short)0, acc1[j], false, false);
    }
  }
  // C/D layout: VGPR r, lanes<16 -> row r, lanes>=16 -> row r+8; column = lane&15
#pragma unroll
  for (int j = 0; j < PPB; ++j) {
#pragma unroll
    for (int r = 0; r < 8; ++r) {
      av0s[r + 8 * kh][wv * 64 + j * 16 + cl] = acc0[j][r];
      av1s[r + 8 * kh][wv * 64 + j * 16 + cl] = acc1[j][r];
    }
  }
  __syncthreads();
  // Epilogue: out[s,pos,o] = av0·M0[o,:] + av1·M1[o,:] + b[o]; M reused over 4 pos.
  int row = tid >> 3;             // 0..15
  int o0  = (tid & 7) * 8;        // 8 outputs per thread per pos, coalesced rows
  float res[PPB][8];
#pragma unroll
  for (int pp = 0; pp < PPB; ++pp)
#pragma unroll
    for (int i = 0; i < 8; ++i) res[pp][i] = fcb[o0 + i];
  for (int d = 0; d < 64; ++d) {
    float m0v[8], m1v[8];
#pragma unroll
    for (int i = 0; i < 8; ++i) {
      m0v[i] = M0[(o0 + i) * 64 + d];
      m1v[i] = M1[(o0 + i) * 64 + d];
    }
#pragma unroll
    for (int pp = 0; pp < PPB; ++pp) {
      float x0 = av0s[row][pp * 64 + d];
      float x1 = av1s[row][pp * 64 + d];
#pragma unroll
      for (int i = 0; i < 8; ++i)
        res[pp][i] += x0 * m0v[i] + x1 * m1v[i];
    }
  }
#pragma unroll
  for (int pp = 0; pp < PPB; ++pp) {
    float* orow =
        out + (((size_t)b * NTOK + st * 16 + row) * NPOS + pos0 + pp) * 64 + o0;
#pragma unroll
    for (int i = 0; i < 8; ++i) orow[i] = res[pp][i];
  }
}

// ---------------------------------------------------------------------------
extern "C" void kernel_launch(void* const* d_in, const int* in_sizes, int n_in,
                              void* d_out, int out_size, void* d_ws, size_t ws_size,
                              hipStream_t stream) {
  const float* q    = (const float*)d_in[0];
  const float* k    = (const float*)d_in[1];
  const float* v    = (const float*)d_in[2];
  const int*   mask = (const int*)d_in[3];
  const float* wq0  = (const float*)d_in[4];
  const float* wk0  = (const float*)d_in[5];
  const float* wv0  = (const float*)d_in[6];
  const float* wq1  = (const float*)d_in[7];
  const float* wk1  = (const float*)d_in[8];
  const float* wv1  = (const float*)d_in[9];
  const float* fc_w = (const float*)d_in[10];
  const float* fc_b = (const float*)d_in[11];
  float* out = (float*)d_out;

  char* ws = (char*)d_ws;
  size_t off = 0;
  const size_t QK = (size_t)BATCH * NTOK * DM * sizeof(float);        // 917504
  float* q00 = (float*)(ws + off); off += QK;
  float* q01 = (float*)(ws + off); off += QK;
  float* k00 = (float*)(ws + off); off += QK;
  float* k01 = (float*)(ws + off); off += QK;
  float* M0  = (float*)(ws + off); off += 64 * 64 * sizeof(float);
  float* M1  = (float*)(ws + off); off += 64 * 64 * sizeof(float);
  const size_t AS = (size_t)BATCH * NTOK * NTOK * sizeof(uint16_t);   // 3211264
  uint16_t* a0 = (uint16_t*)(ws + off); off += AS;
  uint16_t* a1 = (uint16_t*)(ws + off); off += AS;
  uint16_t* vt = (uint16_t*)(ws + off); off += (size_t)BATCH * FDIM * NTOK * sizeof(uint16_t);

  proj_kernel<<<BATCH * NTOK, 64, 0, stream>>>(q, wq0, wq1, q00, q01);
  proj_kernel<<<BATCH * NTOK, 64, 0, stream>>>(k, wk0, wk1, k00, k01);
  mproj_kernel<<<16, 256, 0, stream>>>(fc_w, wv0, wv1, M0, M1);
  vpack_kernel<<<dim3(FDIM / 32, NTOK / 32, BATCH), dim3(32, 8), 0, stream>>>(v, vt);
  scores_kernel<<<BATCH * NTOK, 256, 0, stream>>>(q00, q01, k00, k01, mask, a0, a1);
  attnmm_kernel<<<dim3(NTOK / 16, NPOS / PPB, BATCH), 128, 0, stream>>>(
      a0, a1, vt, M0, M1, fc_b, out);
}
